// SelfJoinLayerWithRetrieval_34153579938034
// MI455X (gfx1250) — matrix operand order repaired and checked by reading, not compiled, not run on hardware
//
#include <hip/hip_runtime.h>

// Problem sizes (fixed by the reference)
#define BN 2048
#define MN 65536
#define CN 128
#define NSL 16                // M-slices for the flash pass
#define SLLEN (MN / NSL)
#define NEGV (-1.0e30f)

typedef __bf16 bf16_t;
typedef __attribute__((ext_vector_type(16))) __bf16 v16bf;
typedef __attribute__((ext_vector_type(8)))  __bf16 v8bf;
typedef __attribute__((ext_vector_type(8)))  float  v8f;
typedef int v4i __attribute__((vector_size(16)));   // matches builtin param type

#if __has_builtin(__builtin_amdgcn_global_load_async_to_lds_b128) && \
    __has_builtin(__builtin_amdgcn_s_wait_asynccnt)
#define USE_ASYNC 1
#else
#define USE_ASYNC 0
#endif

#if defined(__AMDGCN__)
typedef __attribute__((address_space(1))) v4i g_v4i;
typedef __attribute__((address_space(3))) v4i l_v4i;
#endif

// 16-byte global->LDS copy; async (ASYNCcnt-tracked) when available.
static __device__ inline void cp16_g2l(const bf16_t* gp, bf16_t* lp) {
#if USE_ASYNC
  __builtin_amdgcn_global_load_async_to_lds_b128(
      (g_v4i*)const_cast<bf16_t*>(gp), (l_v4i*)lp, 0, 0);
#else
  *(v8bf*)lp = *(const v8bf*)gp;
#endif
}
static __device__ inline void wait_async_le(int n) {
#if USE_ASYNC
  if (n == 0) __builtin_amdgcn_s_wait_asynccnt(0);
  else        __builtin_amdgcn_s_wait_asynccnt(8);
#else
  (void)n;
#endif
}

static __device__ inline v8f wmma_bf16(v16bf a, v16bf b, v8f c) {
  // D = A(16x32) * B(32x16) + C, f32 accumulate
  return __builtin_amdgcn_wmma_f32_16x16x32_bf16(false, a, false, b, (short)0, c, false, false);
}

// A-fragment (16x32 bf16), contiguous k order (not ISA table order): lane holds row (lane&15);
// element e holds K = (lane>>4)*16 + e.
static __device__ inline v16bf afrag_ld(const bf16_t* rowp, int k0, int half) {
  v8bf g0 = *(const v8bf*)(rowp + k0 + half * 16);
  v8bf g1 = *(const v8bf*)(rowp + k0 + half * 16 + 8);
  v16bf r;
#pragma unroll
  for (int e = 0; e < 8; ++e) { r[e] = g0[e]; r[e + 8] = g1[e]; }
  return r;
}

static __device__ inline v16bf afrag_cvt(const float* rowp, int k0, int half, float s) {
  const float* g0 = rowp + k0 + half * 16;
  const float* g1 = rowp + k0 + half * 16 + 8;
  v16bf r;
#pragma unroll
  for (int e = 0; e < 8; ++e) { r[e] = (__bf16)(g0[e] * s); r[e + 8] = (__bf16)(g1[e] * s); }
  return r;
}

// B-fragment (32x16 bf16) from N-major storage: lane holds col (lane&15);
// element e holds K = k0 + (lane>>4)*16 + e  -> 16 contiguous bf16 per lane.
static __device__ inline v16bf bfrag_ld(const bf16_t* p) { return *(const v16bf*)p; }

// ---------------------------------------------------------------- kernel W
// Weight prep: bf16 N-major transposes + ymsg[2][128] = y_emb@W3 + msg_b
__global__ void kW(const float* key_w, const float* msg_w, const float* msg_b,
                   const float* upd_w1, const float* upd_w2, const float* y_emb,
                   bf16_t* keyT, bf16_t* w1T, bf16_t* w2T,
                   bf16_t* u1T, bf16_t* u2T, float* ymsgb) {
  int tid = threadIdx.x;
  for (int i = tid; i < CN * CN; i += 256) {
    int n = i >> 7, k = i & 127;
    keyT[i] = (__bf16)key_w[k * CN + n];
    w1T[i]  = (__bf16)msg_w[k * CN + n];                // W1 = msg_w[:C]
    w2T[i]  = (__bf16)msg_w[(CN + k) * CN + n];         // W2 = msg_w[C:2C]
    u2T[i]  = (__bf16)upd_w2[k * CN + n];
  }
  for (int i = tid; i < CN * 2 * CN; i += 256) {
    int n = i >> 8, k = i & 255;
    u1T[i] = (__bf16)upd_w1[k * CN + n];
  }
  if (tid < 256) {
    int yy = tid >> 7, ch = tid & 127;
    float s = msg_b[ch];
    for (int k = 0; k < CN; ++k) s += y_emb[yy * CN + k] * msg_w[(2 * CN + k) * CN + ch];
    ymsgb[tid] = s;                                     // ymsg + msg_b folded
  }
}

// ---------------------------------------------------------------- kernel M
// Memory prep: mem_feat = l2norm(memory_x); K=mem_feat@key_w+key_b (bf16,(M,C));
// b = mem_feat@W2 + ymsg[y]  stored TRANSPOSED bf16 (C,M).
__global__ __launch_bounds__(128) void kM(const float* mem_x, const int* mem_y,
                   const bf16_t* keyT, const float* key_b, const bf16_t* w2T,
                   const float* ymsgb, bf16_t* k_out, bf16_t* bT) {
  int wid = threadIdx.x >> 5, lane = threadIdx.x & 31;
  int half = lane >> 4, l16 = lane & 15;
  int row0 = (blockIdx.x * 4 + wid) * 16;
  const float* rowp = mem_x + (size_t)(row0 + l16) * CN;

  float ss = 0.f;
#pragma unroll
  for (int j = 0; j < 4; ++j) {
    const float* g0 = rowp + j * 32 + half * 8;
    const float* g1 = rowp + j * 32 + 16 + half * 8;
#pragma unroll
    for (int e = 0; e < 8; ++e) ss += g0[e] * g0[e] + g1[e] * g1[e];
  }
  ss += __shfl_xor(ss, 16);
  float inv = 1.f / fmaxf(sqrtf(ss), 1e-12f);

  v16bf af[4];
#pragma unroll
  for (int j = 0; j < 4; ++j) af[j] = afrag_cvt(rowp, j * 32, half, inv);

  int yr[8];
#pragma unroll
  for (int v = 0; v < 8; ++v) yr[v] = mem_y[row0 + v + 8 * half];

#pragma unroll
  for (int nt = 0; nt < 8; ++nt) {
    int ch = nt * 16 + l16;
    float kb = key_b[ch];
    v8f acc;
#pragma unroll
    for (int v = 0; v < 8; ++v) acc[v] = kb;
#pragma unroll
    for (int j = 0; j < 4; ++j)
      acc = wmma_bf16(af[j], bfrag_ld(keyT + (size_t)ch * CN + j * 32 + half * 16), acc);
#pragma unroll
    for (int v = 0; v < 8; ++v)
      k_out[(size_t)(row0 + v + 8 * half) * CN + ch] = (__bf16)acc[v];

    float y0 = ymsgb[ch], y1 = ymsgb[CN + ch];
    v8f bacc;
#pragma unroll
    for (int v = 0; v < 8; ++v) bacc[v] = yr[v] ? y1 : y0;
#pragma unroll
    for (int j = 0; j < 4; ++j)
      bacc = wmma_bf16(af[j], bfrag_ld(w2T + (size_t)ch * CN + j * 32 + half * 16), bacc);
#pragma unroll
    for (int v = 0; v < 8; ++v)
      bT[(size_t)ch * MN + row0 + v + 8 * half] = (__bf16)bacc[v];
  }
}

// ---------------------------------------------------------------- kernel Q
// q = l2norm(feature)@key_w+key_b (bf16,(B,C));  a = feature@W1 (f32,(B,C))
__global__ __launch_bounds__(128) void kQ(const float* feature, const bf16_t* keyT,
                   const float* key_b, const bf16_t* w1T,
                   bf16_t* q_out, float* a_out) {
  int wid = threadIdx.x >> 5, lane = threadIdx.x & 31;
  int half = lane >> 4, l16 = lane & 15;
  int row0 = (blockIdx.x * 4 + wid) * 16;
  const float* rowp = feature + (size_t)(row0 + l16) * CN;

  float ss = 0.f;
#pragma unroll
  for (int j = 0; j < 4; ++j) {
    const float* g0 = rowp + j * 32 + half * 8;
    const float* g1 = rowp + j * 32 + 16 + half * 8;
#pragma unroll
    for (int e = 0; e < 8; ++e) ss += g0[e] * g0[e] + g1[e] * g1[e];
  }
  ss += __shfl_xor(ss, 16);
  float inv = 1.f / fmaxf(sqrtf(ss), 1e-12f);

  v16bf nf[4], rf[4];
#pragma unroll
  for (int j = 0; j < 4; ++j) {
    nf[j] = afrag_cvt(rowp, j * 32, half, inv);
    rf[j] = afrag_cvt(rowp, j * 32, half, 1.0f);
  }

#pragma unroll
  for (int nt = 0; nt < 8; ++nt) {
    int ch = nt * 16 + l16;
    float kb = key_b[ch];
    v8f qa;
#pragma unroll
    for (int v = 0; v < 8; ++v) qa[v] = kb;
    v8f aa = {};
#pragma unroll
    for (int j = 0; j < 4; ++j) {
      qa = wmma_bf16(nf[j], bfrag_ld(keyT + (size_t)ch * CN + j * 32 + half * 16), qa);
      aa = wmma_bf16(rf[j], bfrag_ld(w1T + (size_t)ch * CN + j * 32 + half * 16), aa);
    }
#pragma unroll
    for (int v = 0; v < 8; ++v) {
      q_out[(size_t)(row0 + v + 8 * half) * CN + ch] = (__bf16)qa[v];
      a_out[(size_t)(row0 + v + 8 * half) * CN + ch] = aa[v];
    }
  }
}

// ---------------------------------------------------------------- kernel F
// Flash pass: 4-wave block = 64 batch rows x one M-slice. Each 32-row chunk of
// K/bT is staged ONCE into LDS with async global->LDS copies (double-buffered,
// ASYNCcnt) and consumed by all 4 waves: 4x cut in L2->WGP traffic.
__global__ __launch_bounds__(128) void kF(const bf16_t* q_out, const bf16_t* k_out,
                   const bf16_t* bT, const float* seed_time, const float* mem_st,
                   float* pacc, float* pmax, float* psum) {
  __shared__ __align__(32) bf16_t kbuf[2][32 * CN];   // K rows, row-major
  __shared__ __align__(32) bf16_t bbuf[2][CN * 32];   // b channels x 32 mem rows
  __shared__ __align__(32) bf16_t pbuf[4][16 * 32];   // per-wave P transpose
  int t = threadIdx.x;
  int wid = t >> 5, lane = t & 31, half = lane >> 4, l16 = lane & 15;
  int grp = blockIdx.x & 31;        // 32 groups of 64 batch rows
  int s = blockIdx.x >> 5;          // NSL slices
  int row0 = grp * 64 + wid * 16;
  int mbeg = s * SLLEN, mend = mbeg + SLLEN;

  v16bf qf[4];
#pragma unroll
  for (int j = 0; j < 4; ++j) qf[j] = afrag_ld(q_out + (size_t)(row0 + l16) * CN, j * 32, half);

  float st[8], mx[8], sm[8];
#pragma unroll
  for (int v = 0; v < 8; ++v) {
    st[v] = seed_time[row0 + v + 8 * half];
    mx[v] = NEGV; sm[v] = 0.f;
  }
  v8f acc[8];
#pragma unroll
  for (int nt = 0; nt < 8; ++nt) acc[nt] = v8f{};

  // stage chunk m0 into buffer buf: 8 x b128 per thread (ASYNCcnt += 8/wave)
  auto stage = [&](int m0, int buf) {
#pragma unroll
    for (int i = 0; i < 4; ++i) {
      cp16_g2l(k_out + (size_t)m0 * CN + (i * 128 + t) * 8, &kbuf[buf][(i * 128 + t) * 8]);
      cp16_g2l(bT + (size_t)t * MN + m0 + i * 8, &bbuf[buf][t * 32 + i * 8]);
    }
  };

  stage(mbeg, 0);
  for (int m0 = mbeg; m0 < mend; m0 += 32) {
    int cur = ((m0 - mbeg) >> 5) & 1;
    if (m0 + 32 < mend) { stage(m0 + 32, cur ^ 1); wait_async_le(8); }
    else                { wait_async_le(0); }
    __syncthreads();                       // chunk [cur] ready for all waves
    const bf16_t* kb = kbuf[cur];
    const bf16_t* bb = bbuf[cur];

    v8f s0 = {}, s1 = {};
#pragma unroll
    for (int j = 0; j < 4; ++j) {
      s0 = wmma_bf16(qf[j], bfrag_ld(kb + l16 * CN + j * 32 + half * 16), s0);
      s1 = wmma_bf16(qf[j], bfrag_ld(kb + (16 + l16) * CN + j * 32 + half * 16), s1);
    }
    float mst0 = mem_st[m0 + l16];
    float mst1 = mem_st[m0 + 16 + l16];

    float sc[8];
#pragma unroll
    for (int v = 0; v < 8; ++v) {
      float a0 = (st[v] < mst0) ? NEGV : s0[v];
      float a1 = (st[v] < mst1) ? NEGV : s1[v];
      float tv = fmaxf(a0, a1);
      tv = fmaxf(tv, __shfl_xor(tv, 1));
      tv = fmaxf(tv, __shfl_xor(tv, 2));
      tv = fmaxf(tv, __shfl_xor(tv, 4));
      tv = fmaxf(tv, __shfl_xor(tv, 8));
      float mn = fmaxf(mx[v], tv);
      sc[v] = __expf(mx[v] - mn);
      mx[v] = mn;
      float p0 = __expf(a0 - mn);
      float p1 = __expf(a1 - mn);
      s0[v] = p0; s1[v] = p1;
      float rs = p0 + p1;
      rs += __shfl_xor(rs, 1);
      rs += __shfl_xor(rs, 2);
      rs += __shfl_xor(rs, 4);
      rs += __shfl_xor(rs, 8);
      sm[v] = sm[v] * sc[v] + rs;
    }
#pragma unroll
    for (int nt = 0; nt < 8; ++nt)
#pragma unroll
      for (int v = 0; v < 8; ++v) acc[nt][v] *= sc[v];

    // transpose P (16x32, C-layout -> A-layout) via per-wave LDS scratch
#pragma unroll
    for (int v = 0; v < 8; ++v) {
      pbuf[wid][(v + 8 * half) * 32 + l16]      = (__bf16)s0[v];
      pbuf[wid][(v + 8 * half) * 32 + 16 + l16] = (__bf16)s1[v];
    }
    __syncthreads();
    v16bf pf = afrag_ld(&pbuf[wid][l16 * 32], 0, half);

#pragma unroll
    for (int nt = 0; nt < 8; ++nt)
      acc[nt] = wmma_bf16(pf, bfrag_ld(bb + (nt * 16 + l16) * 32 + half * 16), acc[nt]);
    __syncthreads();                       // all reads of [cur] done before reissue
  }

#pragma unroll
  for (int nt = 0; nt < 8; ++nt)
#pragma unroll
    for (int v = 0; v < 8; ++v)
      pacc[((size_t)s * BN + row0 + v + 8 * half) * CN + nt * 16 + l16] = acc[nt][v];
  if (l16 == 0) {
#pragma unroll
    for (int v = 0; v < 8; ++v) {
      pmax[s * BN + row0 + v + 8 * half] = mx[v];
      psum[s * BN + row0 + v + 8 * half] = sm[v];
    }
  }
}

// ---------------------------------------------------------------- kernel C
// Combine slices: hagg = a + (sum_s w_s*acc_s) / (sum_s w_s*sum_s), w_s=exp(m_s-m_g)
__global__ void kC(const float* pacc, const float* pmax, const float* psum,
                   const float* a_out, float* hagg) {
  int idx = blockIdx.x * 256 + threadIdx.x;   // < BN*CN
  int row = idx >> 7, ch = idx & 127;
  float mg = NEGV;
#pragma unroll
  for (int s = 0; s < NSL; ++s) mg = fmaxf(mg, pmax[s * BN + row]);
  float sg = 0.f, num = 0.f;
#pragma unroll
  for (int s = 0; s < NSL; ++s) {
    float w = __expf(pmax[s * BN + row] - mg);
    sg += w * psum[s * BN + row];
    num += w * pacc[((size_t)s * BN + row) * CN + ch];
  }
  hagg[idx] = a_out[idx] + num / sg;          // rowsum(softmax)==1 -> 1*a + score@b
}

// ---------------------------------------------------------------- kernel U
// h = relu([feature, hagg] @ upd_w1 + b1);  out = feature + h @ upd_w2 + b2
__global__ __launch_bounds__(32) void kU(const float* feature, const float* hagg,
                   const bf16_t* u1T, const float* ub1,
                   const bf16_t* u2T, const float* ub2, float* out) {
  __shared__ __align__(32) bf16_t lds[16 * CN];
  int lane = threadIdx.x & 31, half = lane >> 4, l16 = lane & 15;
  int row0 = blockIdx.x * 16;
  const float* fr = feature + (size_t)(row0 + l16) * CN;
  const float* hr = hagg + (size_t)(row0 + l16) * CN;

  v16bf ff[4], hf[4];
#pragma unroll
  for (int j = 0; j < 4; ++j) {
    ff[j] = afrag_cvt(fr, j * 32, half, 1.0f);
    hf[j] = afrag_cvt(hr, j * 32, half, 1.0f);
  }

#pragma unroll
  for (int nt = 0; nt < 8; ++nt) {
    int ch = nt * 16 + l16;
    float b1 = ub1[ch];
    v8f acc;
#pragma unroll
    for (int v = 0; v < 8; ++v) acc[v] = b1;
#pragma unroll
    for (int j = 0; j < 4; ++j) {
      acc = wmma_bf16(ff[j], bfrag_ld(u1T + (size_t)ch * 256 + j * 32 + half * 16), acc);
      acc = wmma_bf16(hf[j], bfrag_ld(u1T + (size_t)ch * 256 + 128 + j * 32 + half * 16), acc);
    }
#pragma unroll
    for (int v = 0; v < 8; ++v)
      lds[(v + 8 * half) * CN + ch] = (__bf16)fmaxf(acc[v], 0.f);
  }
  __syncthreads();

  v16bf hh[4];
#pragma unroll
  for (int j = 0; j < 4; ++j) hh[j] = afrag_ld(lds + l16 * CN, j * 32, half);

#pragma unroll
  for (int nt = 0; nt < 8; ++nt) {
    int ch = nt * 16 + l16;
    float b2 = ub2[ch];
    v8f acc;
#pragma unroll
    for (int v = 0; v < 8; ++v) acc[v] = b2;
#pragma unroll
    for (int j = 0; j < 4; ++j)
      acc = wmma_bf16(hh[j], bfrag_ld(u2T + (size_t)ch * CN + j * 32 + half * 16), acc);
#pragma unroll
    for (int v = 0; v < 8; ++v) {
      size_t r = (size_t)(row0 + v + 8 * half);
      out[r * CN + ch] = feature[r * CN + ch] + acc[v];
    }
  }
}

// ----------------------------------------------------------------
extern "C" void kernel_launch(void* const* d_in, const int* in_sizes, int n_in,
                              void* d_out, int out_size, void* d_ws, size_t ws_size,
                              hipStream_t stream) {
  (void)in_sizes; (void)n_in; (void)out_size; (void)ws_size;
  const float* feature  = (const float*)d_in[0];
  const float* memory_x = (const float*)d_in[1];
  const int*   memory_y = (const int*)d_in[2];
  const float* seed_time = (const float*)d_in[3];
  const float* memory_seed_time = (const float*)d_in[4];
  const float* key_w  = (const float*)d_in[5];
  const float* key_b  = (const float*)d_in[6];
  const float* msg_w  = (const float*)d_in[7];
  const float* msg_b  = (const float*)d_in[8];
  const float* upd_w1 = (const float*)d_in[9];
  const float* upd_b1 = (const float*)d_in[10];
  const float* upd_w2 = (const float*)d_in[11];
  const float* upd_b2 = (const float*)d_in[12];
  const float* y_emb  = (const float*)d_in[13];
  float* out = (float*)d_out;

  // workspace carve-up (~53 MB total)
  char* ws = (char*)d_ws;
  size_t off = 0;
  auto take = [&](size_t n) { char* p = ws + off; off = (off + n + 255) & ~(size_t)255; return p; };
  bf16_t* keyT  = (bf16_t*)take((size_t)CN * CN * 2);
  bf16_t* w1T   = (bf16_t*)take((size_t)CN * CN * 2);
  bf16_t* w2T   = (bf16_t*)take((size_t)CN * CN * 2);
  bf16_t* u1T   = (bf16_t*)take((size_t)CN * 2 * CN * 2);
  bf16_t* u2T   = (bf16_t*)take((size_t)CN * CN * 2);
  float*  ymsgb = (float*)take(2 * CN * 4);
  bf16_t* k_out = (bf16_t*)take((size_t)MN * CN * 2);
  bf16_t* bT    = (bf16_t*)take((size_t)CN * MN * 2);
  bf16_t* q_out = (bf16_t*)take((size_t)BN * CN * 2);
  float*  a_out = (float*)take((size_t)BN * CN * 4);
  float*  pacc  = (float*)take((size_t)NSL * BN * CN * 4);
  float*  pmax  = (float*)take((size_t)NSL * BN * 4);
  float*  psum  = (float*)take((size_t)NSL * BN * 4);
  float*  hagg  = (float*)take((size_t)BN * CN * 4);

  kW<<<1, 256, 0, stream>>>(key_w, msg_w, msg_b, upd_w1, upd_w2, y_emb,
                            keyT, w1T, w2T, u1T, u2T, ymsgb);
  kM<<<MN / 16 / 4, 128, 0, stream>>>(memory_x, memory_y, keyT, key_b, w2T, ymsgb, k_out, bT);
  kQ<<<BN / 16 / 4, 128, 0, stream>>>(feature, keyT, key_b, w1T, q_out, a_out);
  kF<<<NSL * (BN / 64), 128, 0, stream>>>(q_out, k_out, bT, seed_time, memory_seed_time,
                                          pacc, pmax, psum);
  kC<<<BN * CN / 256, 256, 0, stream>>>(pacc, pmax, psum, a_out, hagg);
  kU<<<BN / 16, 32, 0, stream>>>(feature, hagg, u1T, upd_b1, u2T, upd_b2, out);
}
